// GraphNetworkBlock_90537910599968
// MI455X (gfx1250) — compile-verified, matrix operand order checked by reading, not run on hardware
//
#include <hip/hip_runtime.h>

typedef __attribute__((ext_vector_type(16))) _Float16 v16h;
typedef __attribute__((ext_vector_type(8)))  _Float16 v8h;
typedef __attribute__((ext_vector_type(8)))  float    v8f;
typedef __attribute__((ext_vector_type(4)))  float    v4f;

#define N_NODES   50000
#define N_EDGES   800000
#define N_GLOB    16
#define DIM       64

// ---------------------------------------------------------------------------
// atomics: non-returning fadd (global agent scope / LDS workgroup scope)
// ---------------------------------------------------------------------------
__device__ __forceinline__ void gatomic_add(float* p, float v) {
  __hip_atomic_fetch_add(p, v, __ATOMIC_RELAXED, __HIP_MEMORY_SCOPE_AGENT);
}
__device__ __forceinline__ void satomic_add(float* p, float v) {
  __hip_atomic_fetch_add(p, v, __ATOMIC_RELAXED, __HIP_MEMORY_SCOPE_WORKGROUP);
}

// ---------------------------------------------------------------------------
// WMMA fragment loaders (CDNA5 16x16x32 f16 layouts, ISA 7.12.2)
// A (16x32, 16-bit): lane<16 -> row=lane, K in {kt*32+0..7, kt*32+16..23}
//                    lane>=16 -> row=lane-16, K in {+8..15, +24..31}
// B (32x16, 16-bit): packed offline into per-lane 16-half runs (same K map).
// ---------------------------------------------------------------------------
__device__ __forceinline__ v16h load_a_frag(const _Float16* act, int ldk, int kt) {
  const int lane = threadIdx.x & 31;
  const int r    = lane & 15;
  const int grp  = lane >> 4;
  const _Float16* p = act + r * ldk + kt * 32 + grp * 8;
  v8h lo = *(const v8h*)p;
  v8h hi = *(const v8h*)(p + 16);
  v16h a;
#pragma unroll
  for (int i = 0; i < 8; ++i) { a[i] = lo[i]; a[i + 8] = hi[i]; }
  return a;
}

__device__ __forceinline__ v16h load_b_frag(const _Float16* frag) {
  const int lane = threadIdx.x & 31;
  const _Float16* p = frag + lane * 16;
  v8h lo = *(const v8h*)p;
  v8h hi = *(const v8h*)(p + 8);
  v16h b;
#pragma unroll
  for (int i = 0; i < 8; ++i) { b[i] = lo[i]; b[i + 8] = hi[i]; }
  return b;
}

// ---------------------------------------------------------------------------
// Dual-M-tile layer: 32 rows, each B fragment feeds two independent WMMA
// accumulator chains (c0/c1) -> halves weight traffic, removes D->C RAW
// serialization between consecutive WMMAs.
// ---------------------------------------------------------------------------
template<int NUMKT, int NUMNT>
__device__ __forceinline__ void mlp_layer_dual(const _Float16* actIn, int ldk,
                                               const _Float16* __restrict__ W,
                                               const float* __restrict__ bias,
                                               _Float16* actOut, int ldo) {
  const int lane = threadIdx.x & 31;
  const int r    = lane & 15;
  const int mo   = (lane >> 4) * 8;
#pragma unroll
  for (int nt = 0; nt < NUMNT; ++nt) {
    v8f c0 = {}, c1 = {};
#pragma unroll
    for (int kt = 0; kt < NUMKT; ++kt) {
      v16h B  = load_b_frag(W + (nt * NUMKT + kt) * 512);
      v16h A0 = load_a_frag(actIn, ldk, kt);
      c0 = __builtin_amdgcn_wmma_f32_16x16x32_f16(false, A0, false, B,
                                                  (short)0, c0, false, false);
      v16h A1 = load_a_frag(actIn + 16 * ldk, ldk, kt);
      c1 = __builtin_amdgcn_wmma_f32_16x16x32_f16(false, A1, false, B,
                                                  (short)0, c1, false, false);
    }
    float bn = bias[nt * 16 + r];   // C layout: column == lane -> scalar bias
#pragma unroll
    for (int v = 0; v < 8; ++v) {
      float a = c0[v] + bn; a = a > 0.f ? a : 0.f;
      float b = c1[v] + bn; b = b > 0.f ? b : 0.f;
      actOut[(mo + v) * ldo + nt * 16 + r]        = (_Float16)a;
      actOut[(16 + mo + v) * ldo + nt * 16 + r]   = (_Float16)b;
    }
  }
}

// Final dual layer (out=64): relu result as f32 (outF has row stride ldoF)
template<int NUMKT>
__device__ __forceinline__ void mlp_layer_dual_last(const _Float16* actIn, int ldk,
                                                    const _Float16* __restrict__ W,
                                                    const float* __restrict__ bias,
                                                    float* outF, int ldoF) {
  const int lane = threadIdx.x & 31;
  const int r    = lane & 15;
  const int mo   = (lane >> 4) * 8;
#pragma unroll
  for (int nt = 0; nt < 4; ++nt) {
    v8f c0 = {}, c1 = {};
#pragma unroll
    for (int kt = 0; kt < NUMKT; ++kt) {
      v16h B  = load_b_frag(W + (nt * NUMKT + kt) * 512);
      v16h A0 = load_a_frag(actIn, ldk, kt);
      c0 = __builtin_amdgcn_wmma_f32_16x16x32_f16(false, A0, false, B,
                                                  (short)0, c0, false, false);
      v16h A1 = load_a_frag(actIn + 16 * ldk, ldk, kt);
      c1 = __builtin_amdgcn_wmma_f32_16x16x32_f16(false, A1, false, B,
                                                  (short)0, c1, false, false);
    }
    float bn = bias[nt * 16 + r];
#pragma unroll
    for (int v = 0; v < 8; ++v) {
      float a = c0[v] + bn; a = a > 0.f ? a : 0.f;
      float b = c1[v] + bn; b = b > 0.f ? b : 0.f;
      outF[(mo + v) * ldoF + nt * 16 + r]      = a;
      outF[(16 + mo + v) * ldoF + nt * 16 + r] = b;
    }
  }
}

// Single-tile variants (global model, 16 rows)
template<int NUMKT, int NUMNT>
__device__ __forceinline__ void mlp_layer_f16(const _Float16* actIn, int ldk,
                                              const _Float16* __restrict__ W,
                                              const float* __restrict__ bias,
                                              _Float16* actOut, int ldo) {
  const int lane = threadIdx.x & 31;
  const int r    = lane & 15;
  const int mo   = (lane >> 4) * 8;
#pragma unroll
  for (int nt = 0; nt < NUMNT; ++nt) {
    v8f c = {};
#pragma unroll
    for (int kt = 0; kt < NUMKT; ++kt) {
      v16h A = load_a_frag(actIn, ldk, kt);
      v16h B = load_b_frag(W + (nt * NUMKT + kt) * 512);
      c = __builtin_amdgcn_wmma_f32_16x16x32_f16(false, A, false, B,
                                                 (short)0, c, false, false);
    }
    float bn = bias[nt * 16 + r];
#pragma unroll
    for (int v = 0; v < 8; ++v) {
      float val = c[v] + bn;
      val = val > 0.f ? val : 0.f;
      actOut[(mo + v) * ldo + nt * 16 + r] = (_Float16)val;
    }
  }
}

template<int NUMKT>
__device__ __forceinline__ void mlp_layer_last(const _Float16* actIn, int ldk,
                                               const _Float16* __restrict__ W,
                                               const float* __restrict__ bias,
                                               float* outF) {
  const int lane = threadIdx.x & 31;
  const int r    = lane & 15;
  const int mo   = (lane >> 4) * 8;
#pragma unroll
  for (int nt = 0; nt < 4; ++nt) {
    v8f c = {};
#pragma unroll
    for (int kt = 0; kt < NUMKT; ++kt) {
      v16h A = load_a_frag(actIn, ldk, kt);
      v16h B = load_b_frag(W + (nt * NUMKT + kt) * 512);
      c = __builtin_amdgcn_wmma_f32_16x16x32_f16(false, A, false, B,
                                                 (short)0, c, false, false);
    }
    float bn = bias[nt * 16 + r];
#pragma unroll
    for (int v = 0; v < 8; ++v) {
      float val = c[v] + bn;
      val = val > 0.f ? val : 0.f;
      outF[(mo + v) * 64 + nt * 16 + r] = val;
    }
  }
}

// Row-wise LayerNorm over 32 rows (one row per lane), rows register-resident
__device__ __forceinline__ void layernorm32(float* base, int ld,
                                            const float* __restrict__ gamma,
                                            const float* __restrict__ beta) {
  const int lane = threadIdx.x & 31;
  float* row = base + lane * ld;
  v4f v[16];
#pragma unroll
  for (int q = 0; q < 16; ++q) v[q] = *(const v4f*)(row + q * 4);
  float s = 0.f;
#pragma unroll
  for (int q = 0; q < 16; ++q) s += v[q][0] + v[q][1] + v[q][2] + v[q][3];
  float mu = s * (1.0f / 64.0f);
  float var = 0.f;
#pragma unroll
  for (int q = 0; q < 16; ++q) {
#pragma unroll
    for (int i = 0; i < 4; ++i) { float d = v[q][i] - mu; var += d * d; }
  }
  float rstd = rsqrtf(var * (1.0f / 64.0f) + 1e-5f);
#pragma unroll
  for (int q = 0; q < 16; ++q) {
#pragma unroll
    for (int i = 0; i < 4; ++i)
      v[q][i] = (v[q][i] - mu) * rstd * gamma[q * 4 + i] + beta[q * 4 + i];
    *(v4f*)(row + q * 4) = v[q];
  }
}

// 16-row LayerNorm (global model)
__device__ __forceinline__ void layernorm_tile(float* outF,
                                               const float* __restrict__ gamma,
                                               const float* __restrict__ beta) {
  const int lane = threadIdx.x & 31;
  if (lane < 16) {
    float* row = outF + lane * 64;
    float s = 0.f;
#pragma unroll
    for (int j = 0; j < 64; ++j) s += row[j];
    float mu = s * (1.0f / 64.0f);
    float var = 0.f;
#pragma unroll
    for (int j = 0; j < 64; ++j) { float d = row[j] - mu; var += d * d; }
    float rstd = rsqrtf(var * (1.0f / 64.0f) + 1e-5f);
#pragma unroll
    for (int j = 0; j < 64; ++j)
      row[j] = (row[j] - mu) * rstd * gamma[j] + beta[j];
  }
}

// ---------------------------------------------------------------------------
// Weight pack: f32 row-major W[K][N] -> f16 WMMA-B fragments
// ---------------------------------------------------------------------------
__global__ void pack_weights_kernel(const float* __restrict__ W, int K, int N,
                                    _Float16* __restrict__ dst) {
  int t = blockIdx.x * 256 + threadIdx.x;
  if (t >= K * N) return;
  int k  = t / N, n = t % N;
  int kt = k >> 5, kl = k & 31;
  int nt = n >> 4, nl = n & 15;
  int numKt = K >> 5;
  int lane = nl + (((kl >> 3) & 1) << 4);
  int i    = (kl & 7) | (((kl >> 4) & 1) << 3);
  dst[(size_t)(nt * numKt + kt) * 512 + lane * 16 + i] = (_Float16)W[t];
}

// ---------------------------------------------------------------------------
// Edge model: 32 edges / wave. act0 cols 128..255 are dead after layer 1 and
// are re-used as the f32 output tile (row stride 128 floats).
// ---------------------------------------------------------------------------
__global__ __launch_bounds__(64)
void edge_mlp_kernel(const float* __restrict__ edge_attr,
                     const float* __restrict__ node_attr,
                     const float* __restrict__ global_attr,
                     const int*   __restrict__ edge_index,
                     const int*   __restrict__ batch,
                     const _Float16* __restrict__ Wp,
                     const float* __restrict__ b1, const float* __restrict__ b2,
                     const float* __restrict__ b3, const float* __restrict__ b4,
                     const float* __restrict__ gamma, const float* __restrict__ beta,
                     float* __restrict__ edge_new,
                     float* __restrict__ recv_agg,
                     float* __restrict__ e2g) {
  __shared__ __align__(16) _Float16 act0[2][32][256];
  __shared__ __align__(16) _Float16 actM[2][32][128];
  __shared__ float accG[N_GLOB][64];
  __shared__ int idxR[2][32], idxC[2][32], idxB[2][32];

  const int w    = threadIdx.x >> 5;
  const int lane = threadIdx.x & 31;

  for (int i = threadIdx.x; i < N_GLOB * 64; i += 64) ((float*)accG)[i] = 0.f;
  __syncthreads();

  const int numM = N_EDGES / 32;
  for (int m = blockIdx.x * 2 + w; m < numM; m += gridDim.x * 2) {
    const int e0 = m * 32;
    {
      int e  = e0 + lane;
      int rw = edge_index[e];
      idxR[w][lane] = rw;
      idxC[w][lane] = edge_index[N_EDGES + e];
      idxB[w][lane] = batch[rw];
    }
    // gather + f32->f16 into [32][256]
    const int seg = lane >> 3;
    const int off = (lane & 7) * 8;
#pragma unroll 4
    for (int r = 0; r < 32; ++r) {
      const float* src;
      if (seg == 0)      src = node_attr  + (size_t)idxC[w][r] * DIM;   // receiver
      else if (seg == 1) src = node_attr  + (size_t)idxR[w][r] * DIM;   // sender
      else if (seg == 2) src = edge_attr  + (size_t)(e0 + r) * DIM;
      else               src = global_attr + (size_t)idxB[w][r] * DIM;  // g2e
      v4f f0 = *(const v4f*)(src + off);
      v4f f1 = *(const v4f*)(src + off + 4);
      v8h h;
#pragma unroll
      for (int i = 0; i < 4; ++i) { h[i] = (_Float16)f0[i]; h[i + 4] = (_Float16)f1[i]; }
      *(v8h*)&act0[w][r][seg * 64 + off] = h;
    }
    // 256 -> 128 -> 128 -> 128 -> 64
    mlp_layer_dual<8, 8>(&act0[w][0][0], 256, Wp,             b1, &actM[w][0][0], 128);
    mlp_layer_dual<4, 8>(&actM[w][0][0], 128, Wp + 64  * 512, b2, &act0[w][0][0], 256);
    mlp_layer_dual<4, 8>(&act0[w][0][0], 256, Wp + 96  * 512, b3, &actM[w][0][0], 128);
    float* outF = (float*)&act0[w][0][128];       // row stride 128 floats
    mlp_layer_dual_last<4>(&actM[w][0][0], 128, Wp + 128 * 512, b4, outF, 128);
    layernorm32(outF, 128, gamma, beta);

    // tile store (all 32 lanes, 2 rows per iteration)
    float* dst = edge_new + (size_t)e0 * DIM;
#pragma unroll
    for (int q = 0; q < 16; ++q) {
      int row = q * 2 + (lane >> 4);
      int col = (lane & 15) * 4;
      *(v4f*)(dst + row * 64 + col) = *(const v4f*)(outF + row * 128 + col);
    }
    // scatter: recv_agg[col] += edge_new ; accG[batch[row]] += edge_new
#pragma unroll 2
    for (int r = 0; r < 32; ++r) {
      float v0 = outF[r * 128 + lane * 2];
      float v1 = outF[r * 128 + lane * 2 + 1];
      float* p = recv_agg + (size_t)idxC[w][r] * DIM + lane * 2;
      gatomic_add(p,     v0);
      gatomic_add(p + 1, v1);
      int b = idxB[w][r];
      satomic_add(&accG[b][lane * 2],     v0);
      satomic_add(&accG[b][lane * 2 + 1], v1);
    }
  }
  __syncthreads();
  for (int i = threadIdx.x; i < N_GLOB * 64; i += 64)
    gatomic_add(e2g + i, ((float*)accG)[i]);
}

// ---------------------------------------------------------------------------
// Node model: 32 nodes / wave (last pair is 16 valid rows; junk rows are
// computed with clamped gathers and never stored).
// ---------------------------------------------------------------------------
__global__ __launch_bounds__(64)
void node_mlp_kernel(const float* __restrict__ node_attr,
                     const float* __restrict__ global_attr,
                     const int*   __restrict__ batch,
                     const float* __restrict__ recv_agg,
                     const _Float16* __restrict__ Wp,
                     const float* __restrict__ b1, const float* __restrict__ b2,
                     const float* __restrict__ b3, const float* __restrict__ b4,
                     const float* __restrict__ gamma, const float* __restrict__ beta,
                     float* __restrict__ node_new,
                     float* __restrict__ n2g) {
  __shared__ __align__(16) _Float16 act0[2][32][256];
  __shared__ __align__(16) _Float16 actM[2][32][128];
  __shared__ float accG[N_GLOB][64];
  __shared__ int idxB[2][32];

  const int w    = threadIdx.x >> 5;
  const int lane = threadIdx.x & 31;

  for (int i = threadIdx.x; i < N_GLOB * 64; i += 64) ((float*)accG)[i] = 0.f;
  __syncthreads();

  const int numM = (N_NODES + 31) / 32;   // 1563, last has 16 valid rows
  for (int m = blockIdx.x * 2 + w; m < numM; m += gridDim.x * 2) {
    const int n0 = m * 32;
    {
      int nn = n0 + lane;
      if (nn >= N_NODES) nn = N_NODES - 1;
      idxB[w][lane] = batch[nn];
    }
    const int seg = lane >> 3;
    const int off = (lane & 7) * 8;
    if (lane < 24) {
#pragma unroll 4
      for (int r = 0; r < 32; ++r) {
        int nr = n0 + r;
        if (nr >= N_NODES) nr = N_NODES - 1;
        const float* src;
        if (seg == 0)      src = node_attr  + (size_t)nr * DIM;
        else if (seg == 1) src = recv_agg   + (size_t)nr * DIM;
        else               src = global_attr + (size_t)idxB[w][r] * DIM;
        v4f f0 = *(const v4f*)(src + off);
        v4f f1 = *(const v4f*)(src + off + 4);
        v8h h;
#pragma unroll
        for (int i = 0; i < 4; ++i) { h[i] = (_Float16)f0[i]; h[i + 4] = (_Float16)f1[i]; }
        *(v8h*)&act0[w][r][seg * 64 + off] = h;
      }
    }
    // 192 -> 128 -> 128 -> 128 -> 64
    mlp_layer_dual<6, 8>(&act0[w][0][0], 256, Wp,             b1, &actM[w][0][0], 128);
    mlp_layer_dual<4, 8>(&actM[w][0][0], 128, Wp + 48  * 512, b2, &act0[w][0][0], 256);
    mlp_layer_dual<4, 8>(&act0[w][0][0], 256, Wp + 80  * 512, b3, &actM[w][0][0], 128);
    float* outF = (float*)&act0[w][0][128];
    mlp_layer_dual_last<4>(&actM[w][0][0], 128, Wp + 112 * 512, b4, outF, 128);
    layernorm32(outF, 128, gamma, beta);

    const int rv = (N_NODES - n0) < 32 ? (N_NODES - n0) : 32;
    float* dst = node_new + (size_t)n0 * DIM;
#pragma unroll
    for (int q = 0; q < 16; ++q) {
      int row = q * 2 + (lane >> 4);
      int col = (lane & 15) * 4;
      if (row < rv)
        *(v4f*)(dst + row * 64 + col) = *(const v4f*)(outF + row * 128 + col);
    }
#pragma unroll 2
    for (int r = 0; r < 32; ++r) {
      if (r < rv) {
        int b = idxB[w][r];
        satomic_add(&accG[b][lane * 2],     outF[r * 128 + lane * 2]);
        satomic_add(&accG[b][lane * 2 + 1], outF[r * 128 + lane * 2 + 1]);
      }
    }
  }
  __syncthreads();
  for (int i = threadIdx.x; i < N_GLOB * 64; i += 64)
    gatomic_add(n2g + i, ((float*)accG)[i]);
}

// ---------------------------------------------------------------------------
// Global model: [n2g, e2g, global_attr] (192) -> MLP -> LN (single wave)
// ---------------------------------------------------------------------------
__global__ __launch_bounds__(32)
void global_mlp_kernel(const float* __restrict__ global_attr,
                       const float* __restrict__ n2g,
                       const float* __restrict__ e2g,
                       const _Float16* __restrict__ Wp,
                       const float* __restrict__ b1, const float* __restrict__ b2,
                       const float* __restrict__ b3, const float* __restrict__ b4,
                       const float* __restrict__ gamma, const float* __restrict__ beta,
                       float* __restrict__ global_new) {
  __shared__ __align__(16) _Float16 act0[16][256];
  __shared__ __align__(16) _Float16 actM[16][128];
  __shared__ __align__(16) float    outF[16][64];

  const int lane = threadIdx.x & 31;
  const int seg  = lane >> 3;
  const int off  = (lane & 7) * 8;
  if (lane < 24) {
#pragma unroll 4
    for (int r = 0; r < 16; ++r) {
      const float* src;
      if (seg == 0)      src = n2g + r * DIM;
      else if (seg == 1) src = e2g + r * DIM;
      else               src = global_attr + r * DIM;
      v4f f0 = *(const v4f*)(src + off);
      v4f f1 = *(const v4f*)(src + off + 4);
      v8h h;
#pragma unroll
      for (int i = 0; i < 4; ++i) { h[i] = (_Float16)f0[i]; h[i + 4] = (_Float16)f1[i]; }
      *(v8h*)&act0[r][seg * 64 + off] = h;
    }
  }
  mlp_layer_f16<6, 8>(&act0[0][0], 256, Wp,             b1, &actM[0][0], 128);
  mlp_layer_f16<4, 8>(&actM[0][0], 128, Wp + 48  * 512, b2, &act0[0][0], 256);
  mlp_layer_f16<4, 8>(&act0[0][0], 256, Wp + 80  * 512, b3, &actM[0][0], 128);
  mlp_layer_last<4>  (&actM[0][0], 128, Wp + 112 * 512, b4, &outF[0][0]);
  layernorm_tile(&outF[0][0], gamma, beta);

  const float* sf = &outF[0][0];
#pragma unroll
  for (int q = 0; q < 8; ++q) {
    int idx = q * 128 + lane * 4;
    *(v4f*)(global_new + idx) = *(const v4f*)(sf + idx);
  }
}

// ---------------------------------------------------------------------------
// Host launcher
// ---------------------------------------------------------------------------
extern "C" void kernel_launch(void* const* d_in, const int* in_sizes, int n_in,
                              void* d_out, int out_size, void* d_ws, size_t ws_size,
                              hipStream_t stream) {
  const float* edge_attr   = (const float*)d_in[0];
  const float* node_attr   = (const float*)d_in[1];
  const float* global_attr = (const float*)d_in[2];
  const int*   edge_index  = (const int*)d_in[3];
  const int*   batch       = (const int*)d_in[4];

  const float* eW[4] = {(const float*)d_in[5], (const float*)d_in[7],
                        (const float*)d_in[9], (const float*)d_in[11]};
  const float* eB[4] = {(const float*)d_in[6], (const float*)d_in[8],
                        (const float*)d_in[10], (const float*)d_in[12]};
  const float* eGam = (const float*)d_in[13];
  const float* eBet = (const float*)d_in[14];

  const float* nW[4] = {(const float*)d_in[15], (const float*)d_in[17],
                        (const float*)d_in[19], (const float*)d_in[21]};
  const float* nB[4] = {(const float*)d_in[16], (const float*)d_in[18],
                        (const float*)d_in[20], (const float*)d_in[22]};
  const float* nGam = (const float*)d_in[23];
  const float* nBet = (const float*)d_in[24];

  const float* gW[4] = {(const float*)d_in[25], (const float*)d_in[27],
                        (const float*)d_in[29], (const float*)d_in[31]};
  const float* gB[4] = {(const float*)d_in[26], (const float*)d_in[28],
                        (const float*)d_in[30], (const float*)d_in[32]};
  const float* gGam = (const float*)d_in[33];
  const float* gBet = (const float*)d_in[34];

  // workspace layout
  float* ws       = (float*)d_ws;
  float* recv_agg = ws;                              // 50000*64 f32
  float* e2g      = ws + (size_t)N_NODES * DIM;      // 16*64 f32
  float* n2g      = e2g + N_GLOB * DIM;              // 16*64 f32
  _Float16* wE = (_Float16*)(n2g + N_GLOB * DIM);    // 144*512 f16
  _Float16* wN = wE + 144 * 512;                     // 128*512 f16
  _Float16* wG = wN + 128 * 512;                     // 128*512 f16

  hipMemsetAsync(d_ws, 0,
                 ((size_t)N_NODES * DIM + 2 * N_GLOB * DIM) * sizeof(float),
                 stream);

  auto pack = [&](const float* W, int K, int N, _Float16* dst) {
    int total = K * N;
    pack_weights_kernel<<<(total + 255) / 256, 256, 0, stream>>>(W, K, N, dst);
  };
  pack(eW[0], 256, 128, wE);
  pack(eW[1], 128, 128, wE + 64 * 512);
  pack(eW[2], 128, 128, wE + 96 * 512);
  pack(eW[3], 128, 64,  wE + 128 * 512);
  pack(nW[0], 192, 128, wN);
  pack(nW[1], 128, 128, wN + 48 * 512);
  pack(nW[2], 128, 128, wN + 80 * 512);
  pack(nW[3], 128, 64,  wN + 112 * 512);
  pack(gW[0], 192, 128, wG);
  pack(gW[1], 128, 128, wG + 48 * 512);
  pack(gW[2], 128, 128, wG + 80 * 512);
  pack(gW[3], 128, 64,  wG + 112 * 512);

  float* out_edge = (float*)d_out;
  float* out_node = out_edge + (size_t)N_EDGES * DIM;
  float* out_glob = out_node + (size_t)N_NODES * DIM;

  edge_mlp_kernel<<<2048, 64, 0, stream>>>(
      edge_attr, node_attr, global_attr, edge_index, batch,
      wE, eB[0], eB[1], eB[2], eB[3], eGam, eBet,
      out_edge, recv_agg, e2g);

  node_mlp_kernel<<<782, 64, 0, stream>>>(
      node_attr, global_attr, batch, recv_agg,
      wN, nB[0], nB[1], nB[2], nB[3], nGam, nBet,
      out_node, n2g);

  global_mlp_kernel<<<1, 32, 0, stream>>>(
      global_attr, n2g, e2g,
      wG, gB[0], gB[1], gB[2], gB[3], gGam, gBet,
      out_glob);
}
